// TejaGPT_34918084116939
// MI455X (gfx1250) — matrix operand order repaired and checked by reading, not compiled, not run on hardware
//
#include <hip/hip_runtime.h>
#include <math.h>

// ---------------- model dims (fixed by reference) ----------------
static constexpr int  LB = 2;          // batch
static constexpr int  TT = 1024;       // seq len
static constexpr int  DD = 1024;       // model dim
static constexpr int  HH = 16;         // heads
static constexpr int  HD = 64;         // head dim
static constexpr int  NL = 6;          // layers
static constexpr int  VV = 50257;      // vocab
static constexpr int  MM = LB * TT;    // 2048 token rows

// ---------------- CDNA5 WMMA types ----------------
typedef __bf16 bf16_t;
typedef bf16_t v16bf __attribute__((ext_vector_type(16)));
typedef bf16_t v8bf  __attribute__((ext_vector_type(8)));
typedef float  v8f   __attribute__((ext_vector_type(8)));
typedef unsigned short us8 __attribute__((ext_vector_type(8)));

__device__ __forceinline__ unsigned short f2bf(float f) {
    unsigned int u = __float_as_uint(f);
    u += 0x7FFFu + ((u >> 16) & 1u);      // round-to-nearest-even
    return (unsigned short)(u >> 16);
}

// assemble one 16-element bf16 fragment from two contiguous 8-element LDS runs
__device__ __forceinline__ v16bf pack16(const unsigned short* lo, const unsigned short* hi) {
    v8bf l = *(const v8bf*)lo;
    v8bf h = *(const v8bf*)hi;
    v16bf r;
#pragma unroll
    for (int e = 0; e < 8; ++e) { r[e] = l[e]; r[e + 8] = h[e]; }
    return r;
}

__device__ __forceinline__ v8f wmma_bf16(v16bf a, v16bf b, v8f c) {
    // D = A(16x32 bf16) * B(32x16 bf16) + C(16x16 f32)
    return __builtin_amdgcn_wmma_f32_16x16x32_bf16(false, a, false, b, (short)0, c, false, false);
}

// ---- CDNA5 async global->LDS copy (16B per lane), tracked by ASYNCcnt ----
__device__ __forceinline__ void async_copy16(void* lds_dst, const void* gsrc) {
    // flat LDS addresses are {aperture, offset[31:0]} -> low 32 bits = DS offset
    unsigned int l = (unsigned int)(unsigned long long)lds_dst;
    asm volatile("global_load_async_to_lds_b128 %0, %1, off"
                 :: "v"(l), "v"(gsrc)
                 : "memory");
}
__device__ __forceinline__ void wait_async() {
    asm volatile("s_wait_asynccnt 0x0" ::: "memory");
}

// ---------------- embedding ----------------
__global__ __launch_bounds__(256) void embed_kernel(const int* __restrict__ idx,
                                                    const float* __restrict__ tokW,
                                                    const float* __restrict__ posW,
                                                    float* __restrict__ x) {
    int i = blockIdx.x * 256 + threadIdx.x;        // over MM*DD
    if (i >= MM * DD) return;
    int m = i >> 10;           // row (b*T + t)
    int d = i & 1023;
    int t = m & (TT - 1);
    x[i] = tokW[(size_t)idx[m] * DD + d] + posW[(size_t)t * DD + d];
}

// ---------------- layernorm: f32 in -> bf16 out ----------------
__global__ __launch_bounds__(256) void ln_kernel(const float* __restrict__ x,
                                                 const float* __restrict__ g,
                                                 const float* __restrict__ b,
                                                 unsigned short* __restrict__ out) {
    __shared__ float red[256];
    const int row = blockIdx.x;
    const int tid = threadIdx.x;
    const float* xr = x + (size_t)row * DD;
    float s = 0.f;
    for (int d = tid; d < DD; d += 256) s += xr[d];
    red[tid] = s; __syncthreads();
    for (int o = 128; o > 0; o >>= 1) { if (tid < o) red[tid] += red[tid + o]; __syncthreads(); }
    float mu = red[0] * (1.0f / DD);
    __syncthreads();
    float v = 0.f;
    for (int d = tid; d < DD; d += 256) { float t = xr[d] - mu; v += t * t; }
    red[tid] = v; __syncthreads();
    for (int o = 128; o > 0; o >>= 1) { if (tid < o) red[tid] += red[tid + o]; __syncthreads(); }
    float rstd = rsqrtf(red[0] * (1.0f / DD) + 1e-5f);
    for (int d = tid; d < DD; d += 256)
        out[(size_t)row * DD + d] = f2bf((xr[d] - mu) * rstd * g[d] + b[d]);
}

// ---------------- general WMMA GEMM ----------------
// C[M,N] = act(A_bf16[M,K] @ W + bias) (+resid) ; W is [K,N] f32, or [N,K] f32 if WT.
// Block: 128x128 tile, 256 thr = 8 waves, each wave owns a 32x64 subtile (2x4 WMMA tiles).
// A tile staged via CDNA5 async global->LDS; W tile converted f32->bf16 in the load path.
template <bool WT, bool DOGELU, bool HASBIAS, bool RESID, bool OUTF32, bool OUTBF16>
__global__ __launch_bounds__(256) void gemm_kernel(const unsigned short* __restrict__ A,
                                                   const float* __restrict__ W,
                                                   const float* __restrict__ bias,
                                                   const float* __restrict__ resid,
                                                   float* __restrict__ Cf,
                                                   unsigned short* __restrict__ Cbf,
                                                   int M, int N, int K) {
    alignas(64) __shared__ unsigned short As[128][32];   // [m][k] bf16
    alignas(64) __shared__ unsigned short Bs[128][32];   // [n][k] bf16 (transposed)
    const int tid  = threadIdx.x;
    const int lane = tid & 31;
    const int wave = tid >> 5;
    const int wm   = wave & 3;     // 4 waves along M (32 rows each)
    const int wn   = wave >> 2;    // 2 waves along N (64 cols each)
    const int lr   = lane & 15;
    const int gg   = lane >> 4;
    const int m0   = blockIdx.y * 128;
    const int n0   = blockIdx.x * 128;

    v8f acc[2][4];
#pragma unroll
    for (int r = 0; r < 2; ++r)
#pragma unroll
        for (int c = 0; c < 4; ++c) { v8f z = {}; acc[r][c] = z; }

    for (int k0 = 0; k0 < K; k0 += 32) {
        // stage A tile: 128x32 bf16 via async global->LDS (16B per lane, 2 chunks)
#pragma unroll
        for (int i = 0; i < 2; ++i) {
            int cc  = tid + i * 256;
            int row = cc >> 2;
            int ko  = (cc & 3) << 3;
            async_copy16(&As[row][ko], &A[(size_t)(m0 + row) * K + k0 + ko]);
        }
        if (k0 + 32 < K)  // CDNA5 global_prefetch of next A tile
            __builtin_prefetch(&A[(size_t)(m0 + (tid >> 2)) * K + k0 + 32], 0, 1);
        // stage W tile as bf16, transposed to [n][k]
        if (WT) {
#pragma unroll
            for (int i = 0; i < 4; ++i) {
                int j  = tid + i * 256;
                int n  = j >> 3;
                int kq = (j & 7) << 2;
                float4 w;
                if (n0 + n < N) w = *(const float4*)&W[(size_t)(n0 + n) * K + k0 + kq];
                else { w.x = w.y = w.z = w.w = 0.f; }
                Bs[n][kq + 0] = f2bf(w.x); Bs[n][kq + 1] = f2bf(w.y);
                Bs[n][kq + 2] = f2bf(w.z); Bs[n][kq + 3] = f2bf(w.w);
            }
        } else {
#pragma unroll
            for (int i = 0; i < 4; ++i) {
                int j  = tid + i * 256;
                int kk = j >> 5;
                int nn = (j & 31) << 2;
                const float* src = &W[(size_t)(k0 + kk) * N + n0 + nn];
                float4 w;
                if (n0 + nn + 3 < N) w = *(const float4*)src;
                else {
                    w.x = (n0 + nn + 0 < N) ? src[0] : 0.f;
                    w.y = (n0 + nn + 1 < N) ? src[1] : 0.f;
                    w.z = (n0 + nn + 2 < N) ? src[2] : 0.f;
                    w.w = (n0 + nn + 3 < N) ? src[3] : 0.f;
                }
                Bs[nn + 0][kk] = f2bf(w.x); Bs[nn + 1][kk] = f2bf(w.y);
                Bs[nn + 2][kk] = f2bf(w.z); Bs[nn + 3][kk] = f2bf(w.w);
            }
        }
        wait_async();        // ASYNCcnt -> 0 before the workgroup barrier
        __syncthreads();
        v16bf af[2], bfr[4];
#pragma unroll
        for (int r = 0; r < 2; ++r) {
            int ar = wm * 32 + r * 16 + lr;
            af[r] = pack16(&As[ar][gg * 8], &As[ar][16 + gg * 8]);
        }
#pragma unroll
        for (int c = 0; c < 4; ++c) {
            int bn = wn * 64 + c * 16 + lr;
            bfr[c] = pack16(&Bs[bn][gg * 8], &Bs[bn][16 + gg * 8]);
        }
#pragma unroll
        for (int r = 0; r < 2; ++r)
#pragma unroll
            for (int c = 0; c < 4; ++c)
                acc[r][c] = wmma_bf16(af[r], bfr[c], acc[r][c]);
        __syncthreads();
    }

    // epilogue (C layout: m = tile_base + g*8 + vgpr, n = tile_base + lane%16)
#pragma unroll
    for (int r = 0; r < 2; ++r) {
        int mb = m0 + wm * 32 + r * 16 + gg * 8;
#pragma unroll
        for (int c = 0; c < 4; ++c) {
            int n = n0 + wn * 64 + c * 16 + lr;
            if (n < N) {
#pragma unroll
                for (int e = 0; e < 8; ++e) {
                    int m = mb + e;
                    float v = acc[r][c][e];
                    if (HASBIAS) v += bias[n];
                    if (DOGELU)  v = 0.5f * v * (1.0f + erff(v * 0.70710678118654752f));
                    if (RESID)   v += resid[(size_t)m * N + n];
                    if (OUTF32)  Cf[(size_t)m * N + n] = v;
                    if (OUTBF16) Cbf[(size_t)m * N + n] = f2bf(v);
                }
            }
        }
    }
}

// ---------------- attention: scores = mask(q @ k^T / sqrt(HD)) ----------------
__global__ __launch_bounds__(256) void attn_scores_kernel(const unsigned short* __restrict__ qbf,
                                                          const unsigned short* __restrict__ kbf,
                                                          float* __restrict__ scores) {
    alignas(64) __shared__ unsigned short Qs[128][64];
    alignas(64) __shared__ unsigned short Ks[128][64];
    const int tid = threadIdx.x, lane = tid & 31, wave = tid >> 5;
    const int wm = wave & 3, wn = wave >> 2;
    const int lr = lane & 15, gg = lane >> 4;
    const int bh = blockIdx.z, bb = bh >> 4, hh = bh & 15;
    const int tq0 = blockIdx.y * 128;
    const int tk0 = blockIdx.x * 128;
    const bool anyValid = (tk0 <= tq0 + 127);   // fully-masked blocks skip the math

    v8f acc[2][4];
#pragma unroll
    for (int r = 0; r < 2; ++r)
#pragma unroll
        for (int c = 0; c < 4; ++c) { v8f z = {}; acc[r][c] = z; }

    if (anyValid) {
#pragma unroll
        for (int i = 0; i < 4; ++i) {
            int j   = tid + i * 256;
            int row = j >> 3;
            int ko  = (j & 7) << 3;
            async_copy16(&Qs[row][ko], &qbf[(size_t)(bb * TT + tq0 + row) * DD + hh * 64 + ko]);
            async_copy16(&Ks[row][ko], &kbf[(size_t)(bb * TT + tk0 + row) * DD + hh * 64 + ko]);
        }
        wait_async();
        __syncthreads();
#pragma unroll
        for (int kk = 0; kk < 64; kk += 32) {
            v16bf af[2], bfr[4];
#pragma unroll
            for (int r = 0; r < 2; ++r) {
                int ar = wm * 32 + r * 16 + lr;
                af[r] = pack16(&Qs[ar][kk + gg * 8], &Qs[ar][kk + 16 + gg * 8]);
            }
#pragma unroll
            for (int c = 0; c < 4; ++c) {
                int bn = wn * 64 + c * 16 + lr;
                bfr[c] = pack16(&Ks[bn][kk + gg * 8], &Ks[bn][kk + 16 + gg * 8]);
            }
#pragma unroll
            for (int r = 0; r < 2; ++r)
#pragma unroll
                for (int c = 0; c < 4; ++c)
                    acc[r][c] = wmma_bf16(af[r], bfr[c], acc[r][c]);
        }
    }
#pragma unroll
    for (int r = 0; r < 2; ++r) {
        int mb = tq0 + wm * 32 + r * 16 + gg * 8;
#pragma unroll
        for (int c = 0; c < 4; ++c) {
            int n = tk0 + wn * 64 + c * 16 + lr;
#pragma unroll
            for (int e = 0; e < 8; ++e) {
                int m = mb + e;
                float v = (anyValid && n <= m) ? acc[r][c][e] * 0.125f : -INFINITY;
                scores[((size_t)bh * TT + m) * TT + n] = v;
            }
        }
    }
}

// ---------------- softmax over last dim, in place ----------------
__global__ __launch_bounds__(256) void softmax_kernel(float* __restrict__ scores) {
    __shared__ float red[256];
    const int tid = threadIdx.x;
    float* p = scores + (size_t)blockIdx.x * TT;
    float mx = -INFINITY;
    for (int d = tid; d < TT; d += 256) mx = fmaxf(mx, p[d]);
    red[tid] = mx; __syncthreads();
    for (int o = 128; o > 0; o >>= 1) { if (tid < o) red[tid] = fmaxf(red[tid], red[tid + o]); __syncthreads(); }
    mx = red[0]; __syncthreads();
    float s = 0.f;
    for (int d = tid; d < TT; d += 256) { float e = __expf(p[d] - mx); p[d] = e; s += e; }
    red[tid] = s; __syncthreads();
    for (int o = 128; o > 0; o >>= 1) { if (tid < o) red[tid] += red[tid + o]; __syncthreads(); }
    float inv = 1.0f / red[0];
    for (int d = tid; d < TT; d += 256) p[d] *= inv;
}

// ---------------- attention: att = wei @ v (per batch-head) ----------------
__global__ __launch_bounds__(256) void attn_av_kernel(const float* __restrict__ wei,
                                                      const unsigned short* __restrict__ vbf,
                                                      unsigned short* __restrict__ attbf) {
    alignas(64) __shared__ unsigned short Ws[128][32];  // [q][k]
    alignas(64) __shared__ unsigned short Vs[64][32];   // [hd][k]
    const int tid = threadIdx.x, lane = tid & 31, wave = tid >> 5;
    const int lr = lane & 15, gg = lane >> 4;
    const int bh = blockIdx.z, bb = bh >> 4, hh = bh & 15;
    const int m0 = blockIdx.x * 128;

    v8f acc[4];
#pragma unroll
    for (int c = 0; c < 4; ++c) { v8f z = {}; acc[c] = z; }

    for (int k0 = 0; k0 < TT; k0 += 32) {
#pragma unroll
        for (int i = 0; i < 4; ++i) {
            int j   = tid + i * 256;
            int row = j >> 3;
            int kq  = (j & 7) << 2;
            float4 w = *(const float4*)&wei[((size_t)bh * TT + m0 + row) * TT + k0 + kq];
            Ws[row][kq + 0] = f2bf(w.x); Ws[row][kq + 1] = f2bf(w.y);
            Ws[row][kq + 2] = f2bf(w.z); Ws[row][kq + 3] = f2bf(w.w);
        }
        {
            int k  = tid >> 3;            // 0..31
            int nq = (tid & 7) << 3;      // 0..56
            us8 ch = *(const us8*)&vbf[(size_t)(bb * TT + k0 + k) * DD + hh * 64 + nq];
#pragma unroll
            for (int e = 0; e < 8; ++e) Vs[nq + e][k] = ch[e];
        }
        __syncthreads();
        int ar = wave * 16 + lr;
        v16bf af = pack16(&Ws[ar][gg * 8], &Ws[ar][16 + gg * 8]);
        v16bf bfr[4];
#pragma unroll
        for (int c = 0; c < 4; ++c) {
            int bn = c * 16 + lr;
            bfr[c] = pack16(&Vs[bn][gg * 8], &Vs[bn][16 + gg * 8]);
        }
#pragma unroll
        for (int c = 0; c < 4; ++c) acc[c] = wmma_bf16(af, bfr[c], acc[c]);
        __syncthreads();
    }
#pragma unroll
    for (int c = 0; c < 4; ++c) {
        int n  = c * 16 + lr;
        int mb = m0 + wave * 16 + gg * 8;
#pragma unroll
        for (int e = 0; e < 8; ++e)
            attbf[(size_t)(bb * TT + mb + e) * DD + hh * 64 + n] = f2bf(acc[c][e]);
    }
}

// ---------------- host orchestration ----------------
extern "C" void kernel_launch(void* const* d_in, const int* in_sizes, int n_in,
                              void* d_out, int out_size, void* d_ws, size_t ws_size,
                              hipStream_t stream) {
    (void)in_sizes; (void)n_in; (void)out_size; (void)ws_size;
    const int*   idx  = (const int*)  d_in[0];
    const float* tokW = (const float*)d_in[1];
    const float* posW = (const float*)d_in[2];
    const float* ln1g = (const float*)d_in[3];
    const float* ln1b = (const float*)d_in[4];
    const float* Wq   = (const float*)d_in[5];
    const float* Wk   = (const float*)d_in[6];
    const float* Wv   = (const float*)d_in[7];
    const float* Wo   = (const float*)d_in[8];
    const float* bo   = (const float*)d_in[9];
    const float* ln2g = (const float*)d_in[10];
    const float* ln2b = (const float*)d_in[11];
    const float* W1   = (const float*)d_in[12];
    const float* b1   = (const float*)d_in[13];
    const float* W2   = (const float*)d_in[14];
    const float* b2   = (const float*)d_in[15];
    const float* lnfg = (const float*)d_in[16];
    const float* lnfb = (const float*)d_in[17];

    char* ws = (char*)d_ws;
    size_t off = 0;
    auto alloc = [&](size_t bytes) { void* p = ws + off; off += (bytes + 255) & ~(size_t)255; return p; };
    float*          x      = (float*)         alloc((size_t)MM * DD * 4);          //   8 MB
    unsigned short* hbf    = (unsigned short*)alloc((size_t)MM * DD * 2);          //   4 MB
    unsigned short* qbf    = (unsigned short*)alloc((size_t)MM * DD * 2);
    unsigned short* kbf    = (unsigned short*)alloc((size_t)MM * DD * 2);
    unsigned short* vbf    = (unsigned short*)alloc((size_t)MM * DD * 2);
    unsigned short* attbf  = (unsigned short*)alloc((size_t)MM * DD * 2);
    unsigned short* ffnbf  = (unsigned short*)alloc((size_t)MM * 4 * DD * 2);      //  16 MB
    float*          scores = (float*)         alloc((size_t)LB * HH * TT * TT * 4);// 128 MB (fits L2)

    dim3 blk(256);
    embed_kernel<<<(MM * DD) / 256, blk, 0, stream>>>(idx, tokW, posW, x);

    dim3 gD(DD / 128, MM / 128);          // N=1024 GEMMs
    dim3 gF(4 * DD / 128, MM / 128);      // N=4096 GEMM
    for (int l = 0; l < NL; ++l) {
        const float* wq = Wq + (size_t)l * DD * DD;
        const float* wk = Wk + (size_t)l * DD * DD;
        const float* wv = Wv + (size_t)l * DD * DD;
        const float* wo = Wo + (size_t)l * DD * DD;
        const float* w1 = W1 + (size_t)l * DD * 4 * DD;
        const float* w2 = W2 + (size_t)l * 4 * DD * DD;

        ln_kernel<<<MM, blk, 0, stream>>>(x, ln1g + l * DD, ln1b + l * DD, hbf);
        gemm_kernel<false,false,false,false,false,true><<<gD, blk, 0, stream>>>(hbf, wq, nullptr, nullptr, nullptr, qbf, MM, DD, DD);
        gemm_kernel<false,false,false,false,false,true><<<gD, blk, 0, stream>>>(hbf, wk, nullptr, nullptr, nullptr, kbf, MM, DD, DD);
        gemm_kernel<false,false,false,false,false,true><<<gD, blk, 0, stream>>>(hbf, wv, nullptr, nullptr, nullptr, vbf, MM, DD, DD);

        attn_scores_kernel<<<dim3(TT / 128, TT / 128, LB * HH), blk, 0, stream>>>(qbf, kbf, scores);
        softmax_kernel<<<LB * HH * TT, blk, 0, stream>>>(scores);
        attn_av_kernel<<<dim3(TT / 128, 1, LB * HH), blk, 0, stream>>>(scores, vbf, attbf);

        gemm_kernel<false,false,true,true,true,false><<<gD, blk, 0, stream>>>(attbf, wo, bo + l * DD, x, x, nullptr, MM, DD, DD);
        ln_kernel<<<MM, blk, 0, stream>>>(x, ln2g + l * DD, ln2b + l * DD, hbf);
        gemm_kernel<false,true,true,false,false,true><<<gF, blk, 0, stream>>>(hbf, w1, b1 + (size_t)l * 4 * DD, nullptr, nullptr, ffnbf, MM, 4 * DD, DD);
        gemm_kernel<false,false,true,true,true,false><<<gD, blk, 0, stream>>>(ffnbf, w2, b2 + l * DD, x, x, nullptr, MM, DD, 4 * DD);
    }

    ln_kernel<<<MM, blk, 0, stream>>>(x, lnfg, lnfb, hbf);
    // weight-tied lm_head: logits = h @ tok_W^T  (W stored [V, D] -> WT path)
    gemm_kernel<true,false,false,false,true,false><<<dim3((VV + 127) / 128, MM / 128), blk, 0, stream>>>(
        hbf, tokW, nullptr, nullptr, (float*)d_out, nullptr, MM, VV, DD);
}
